// PinSAGELayer_32341103739251
// MI455X (gfx1250) — matrix operand order repaired and checked by reading, not compiled, not run on hardware
//
#include <hip/hip_runtime.h>

typedef __attribute__((ext_vector_type(16))) _Float16 v16h;
typedef __attribute__((ext_vector_type(8)))  float    v8f;

#define N_NODES 100000
#define N_EDGES 1600000
#define IN_DIM  64
#define OUT_DIM 64

// ---------------------------------------------------------------------------
// Fragment loaders (layouts per CDNA5 ISA §7.12.2, wave32)
// ---------------------------------------------------------------------------

// A: 16x32 f16 tile from an f32 row-major matrix (ld floats per row).
// lane&15 = M row; (lane>>4)*8 selects K octet base; elems 0..7 -> K=kb..kb+7,
// elems 8..15 -> K=kb+16..kb+23. Two 32B-aligned float4 pairs -> b128 loads.
static __device__ __forceinline__ v16h load_a_f32(const float* __restrict__ base,
                                                  int row, int ld, int kk, int lane) {
  const int kb = (lane >> 4) * 8;
  const float* p = base + (long long)row * ld + kk + kb;
  const float4 f0 = *(const float4*)(p);
  const float4 f1 = *(const float4*)(p + 4);
  const float4 f2 = *(const float4*)(p + 16);
  const float4 f3 = *(const float4*)(p + 20);
  v16h a;
  a[0] = (_Float16)f0.x;  a[1] = (_Float16)f0.y;
  a[2] = (_Float16)f0.z;  a[3] = (_Float16)f0.w;
  a[4] = (_Float16)f1.x;  a[5] = (_Float16)f1.y;
  a[6] = (_Float16)f1.z;  a[7] = (_Float16)f1.w;
  a[8]  = (_Float16)f2.x; a[9]  = (_Float16)f2.y;
  a[10] = (_Float16)f2.z; a[11] = (_Float16)f2.w;
  a[12] = (_Float16)f3.x; a[13] = (_Float16)f3.y;
  a[14] = (_Float16)f3.z; a[15] = (_Float16)f3.w;
  return a;
}

// B: weights pre-swizzled into per-fragment, per-lane contiguous 32B chunks.
// frag = (kk/32)*4 + ntile; lane l's 16 halves live at ((frag*32 + l) * 16).
static __device__ __forceinline__ v16h load_b_swz(const _Float16* __restrict__ base,
                                                  int frag, int lane) {
  return *(const v16h*)(base + ((frag * 32 + lane) << 4));
}

// ---------------------------------------------------------------------------
// Kernel 0: convert + swizzle weights f32 -> f16 B-fragment layout.
// Source frag element: row = kstep*32 + (l>>4)*8 + (i&7) + (i>>3)*16, col = ntile*16 + (l&15)
// ---------------------------------------------------------------------------
__global__ void pinsage_pack_weights(const float* __restrict__ qw,
                                     const float* __restrict__ ww,
                                     _Float16* __restrict__ qws,
                                     _Float16* __restrict__ wws) {
  const int o = blockIdx.x * blockDim.x + threadIdx.x;
  const int f = o >> 9;          // fragment index
  const int rem = o & 511;
  const int l = rem >> 4;        // lane
  const int i = rem & 15;        // element within lane
  const int kstep = f >> 2;
  const int t = f & 3;
  const int row = kstep * 32 + ((l >> 4) * 8) + (i & 7) + ((i >> 3) * 16);
  const int col = t * 16 + (l & 15);
  if (o < 8 * 512)  qws[o] = (_Float16)qw[row * OUT_DIM + col];   // q_w: 64x64
  if (o < 16 * 512) wws[o] = (_Float16)ww[row * OUT_DIM + col];   // w_w: 128x64
}

// ---------------------------------------------------------------------------
// Kernel 1: h = ReLU(features @ q_w + q_b)   [N,64] = [N,64]x[64,64]
// 1 wave -> 16 rows x 64 cols; 2 k-steps of 32; 8 WMMAs per wave.
// ---------------------------------------------------------------------------
__global__ __launch_bounds__(256) void pinsage_gemm1(
    const float* __restrict__ feat, const _Float16* __restrict__ qws,
    const float* __restrict__ qb, float* __restrict__ h) {
  const int lane = threadIdx.x & 31;
  const int wave = threadIdx.x >> 5;
  const int rowbase = (blockIdx.x * 8 + wave) * 16;
  if (rowbase >= N_NODES) return;   // wave-uniform: EXEC stays all-ones inside
  const int arow = rowbase + (lane & 15);

  v8f acc[4] = {};
#pragma unroll
  for (int kk = 0; kk < IN_DIM; kk += 32) {
    v16h a = load_a_f32(feat, arow, IN_DIM, kk, lane);
#pragma unroll
    for (int t = 0; t < 4; ++t) {
      v16h b = load_b_swz(qws, (kk >> 5) * 4 + t, lane);
      acc[t] = __builtin_amdgcn_wmma_f32_16x16x32_f16(
          false, a, false, b, (short)0, acc[t], false, false);
    }
  }

#pragma unroll
  for (int t = 0; t < 4; ++t) {
    const int col = t * 16 + (lane & 15);
    const float bias = qb[col];
#pragma unroll
    for (int r = 0; r < 8; ++r) {
      const int row = rowbase + ((lane >> 4) * 8) + r;
      float v = acc[t][r] + bias;
      h[(long long)row * IN_DIM + col] = v > 0.f ? v : 0.f;
    }
  }
}

// ---------------------------------------------------------------------------
// Kernel 2: zero h_n
// ---------------------------------------------------------------------------
__global__ void pinsage_zero(float4* __restrict__ p, int n4) {
  int i = blockIdx.x * blockDim.x + threadIdx.x;
  if (i < n4) p[i] = make_float4(0.f, 0.f, 0.f, 0.f);
}

// ---------------------------------------------------------------------------
// Kernel 3: h_n[dst] += h[src] * alpha
// 16 threads/edge, float4 gather of h (L2-resident), scalar f32 atomics to L2.
// ---------------------------------------------------------------------------
__global__ __launch_bounds__(256) void pinsage_scatter(
    const float* __restrict__ h, const float* __restrict__ alpha,
    const int* __restrict__ src, const int* __restrict__ dst,
    float* __restrict__ hn) {
  long long gid = (long long)blockIdx.x * blockDim.x + threadIdx.x;
  if (gid >= (long long)N_EDGES * (IN_DIM / 4)) return;
  const int e  = (int)(gid >> 4);
  const int d4 = (int)(gid & 15) * 4;
  const float a = alpha[e];
  const float4 hv = *(const float4*)&h[(long long)src[e] * IN_DIM + d4];
  float* dptr = &hn[(long long)dst[e] * IN_DIM + d4];
  atomicAdd(dptr + 0, hv.x * a);
  atomicAdd(dptr + 1, hv.y * a);
  atomicAdd(dptr + 2, hv.z * a);
  atomicAdd(dptr + 3, hv.w * a);
}

// ---------------------------------------------------------------------------
// Kernel 4: z = ReLU([h ‖ h_n] @ w_w + w_b), then L2-normalize rows -> out
// K=128 (4 k-steps of 32, first 2 from h, last 2 from h_n). 16 WMMAs/wave.
// ---------------------------------------------------------------------------
__global__ __launch_bounds__(256) void pinsage_gemm2_norm(
    const float* __restrict__ h, const float* __restrict__ hn,
    const _Float16* __restrict__ wws, const float* __restrict__ wb,
    float* __restrict__ out) {
  __shared__ float zs[8][16][72];   // pad row stride vs 64 banks
  __shared__ float invs[8][16];

  const int lane = threadIdx.x & 31;
  const int wave = threadIdx.x >> 5;
  const int rowbase = (blockIdx.x * 8 + wave) * 16;
  const bool active = rowbase < N_NODES;
  const int rb = active ? rowbase : 0;          // clamp: safe loads, skipped stores
  const int arow = rb + (lane & 15);

  v8f acc[4] = {};
#pragma unroll
  for (int kk = 0; kk < 128; kk += 32) {
    const float* A = (kk < 64) ? h : hn;
    v16h a = load_a_f32(A, arow, IN_DIM, kk & 63, lane);
#pragma unroll
    for (int t = 0; t < 4; ++t) {
      v16h b = load_b_swz(wws, (kk >> 5) * 4 + t, lane);
      acc[t] = __builtin_amdgcn_wmma_f32_16x16x32_f16(
          false, a, false, b, (short)0, acc[t], false, false);
    }
  }

  // bias + ReLU, stage tile to LDS
#pragma unroll
  for (int t = 0; t < 4; ++t) {
    const int col = t * 16 + (lane & 15);
    const float bias = wb[col];
#pragma unroll
    for (int r = 0; r < 8; ++r) {
      float v = acc[t][r] + bias;
      zs[wave][(lane >> 4) * 8 + r][col] = v > 0.f ? v : 0.f;
    }
  }
  __syncthreads();

  // per-row sum of squares: 2 lanes per row, 32 cols each, combine via shuffle
  {
    const int r16 = lane >> 1;
    const int co = (lane & 1) * 32;
    float s = 0.f;
#pragma unroll
    for (int j = 0; j < 32; ++j) {
      float v = zs[wave][r16][co + j];
      s += v * v;
    }
    s += __shfl_xor(s, 1, 32);
    if ((lane & 1) == 0) invs[wave][r16] = (s > 0.f) ? rsqrtf(s) : 0.f;
  }
  __syncthreads();

  if (active) {
#pragma unroll
    for (int r = 0; r < 16; ++r) {
      const float inv = invs[wave][r];
      const long long o = (long long)(rowbase + r) * OUT_DIM;
      out[o + lane]      = zs[wave][r][lane] * inv;
      out[o + 32 + lane] = zs[wave][r][32 + lane] * inv;
    }
  }
}

// ---------------------------------------------------------------------------
// Launch
// ---------------------------------------------------------------------------
extern "C" void kernel_launch(void* const* d_in, const int* in_sizes, int n_in,
                              void* d_out, int out_size, void* d_ws, size_t ws_size,
                              hipStream_t stream) {
  const float* feat  = (const float*)d_in[0];
  const float* qw    = (const float*)d_in[1];
  const float* qb    = (const float*)d_in[2];
  const float* ww    = (const float*)d_in[3];
  const float* wb    = (const float*)d_in[4];
  const float* alpha = (const float*)d_in[5];
  const int*   src   = (const int*)d_in[6];
  const int*   dst   = (const int*)d_in[7];
  float* out = (float*)d_out;

  // workspace carve-up: qws (8KB) | wws (16KB) | h [N,64] f32 | h_n [N,64] f32
  char* ws = (char*)d_ws;
  _Float16* qws = (_Float16*)ws;
  _Float16* wws = (_Float16*)(ws + 8192);
  float* h  = (float*)(ws + 32768);
  float* hn = (float*)(ws + 32768 + (size_t)N_NODES * IN_DIM * sizeof(float));

  pinsage_pack_weights<<<32, 256, 0, stream>>>(qw, ww, qws, wws);

  const int gemmBlocks = (N_NODES + 127) / 128;   // 8 waves/block, 16 rows/wave
  pinsage_gemm1<<<gemmBlocks, 256, 0, stream>>>(feat, qws, qb, h);

  const int n4 = N_NODES * IN_DIM / 4;
  pinsage_zero<<<(n4 + 255) / 256, 256, 0, stream>>>((float4*)hn, n4);

  const long long tot = (long long)N_EDGES * (IN_DIM / 4);
  pinsage_scatter<<<(int)((tot + 255) / 256), 256, 0, stream>>>(h, alpha, src, dst, hn);

  pinsage_gemm2_norm<<<gemmBlocks, 256, 0, stream>>>(h, hn, wws, wb, out);
}
